// GAT_R_to_E_7430293422977
// MI455X (gfx1250) — matrix operand order repaired and checked by reading, not compiled, not run on hardware
//
#include <hip/hip_runtime.h>
#include <hip/hip_bf16.h>

// ---------------------------------------------------------------------------
// GAT relation->entity aggregation for MI455X (gfx1250, wave32).
//   Phase 1: score GEMVs via v_wmma_f32_16x16x32_bf16 (one 16-row tile / wave)
//   Phase 2: segment-max via order-preserving u32 atomicMax
//   Phase 3: segment-sum of exp via f32 HW atomic add
//   Phase 4: weighted scatter of x_r rows (one wave per edge, L2-resident out)
// ---------------------------------------------------------------------------

typedef __attribute__((ext_vector_type(16))) __bf16 bf16x16;
typedef __attribute__((ext_vector_type(8)))  __bf16 bf16x8;
typedef __attribute__((ext_vector_type(8)))  float  floatx8;

#define HID 128

__device__ __forceinline__ float lrelu(float x) { return x > 0.0f ? x : 0.01f * x; }

// order-preserving float <-> unsigned encoding (for atomicMax on floats)
__device__ __forceinline__ unsigned enc_f(float x) {
  unsigned u = __float_as_uint(x);
  return (u & 0x80000000u) ? ~u : (u | 0x80000000u);
}
__device__ __forceinline__ float dec_f(unsigned u) {
  return __uint_as_float((u & 0x80000000u) ? (u ^ 0x80000000u) : ~u);
}

__device__ __forceinline__ void fatomicAdd(float* p, float v) {
  // guaranteed HW global_atomic_add_f32 path (no CAS loop)
  unsafeAtomicAdd(p, v);
}

// ---------------------------------------------------------------------------
// Phase 1: S[:,0] = X @ w0, S[:,1] = X @ w1 with WMMA bf16 (f32 accumulate).
// One wave per 16-row tile; K = 128 -> 4 x v_wmma_f32_16x16x32_bf16.
// All loads are unconditional wide vector loads (b128); partial tiles are
// handled by clamping the row index (dup rows) and guarding only the store.
// ---------------------------------------------------------------------------
__global__ void scores_wmma_kernel(const float* __restrict__ X, int nrows,
                                   const float* __restrict__ w0,
                                   const float* __restrict__ w1,
                                   float* __restrict__ s0,
                                   float* __restrict__ s1,
                                   int ntiles) {
  const int lane = threadIdx.x & 31;
  const int wave = threadIdx.x >> 5;
  const int tile = blockIdx.x * (blockDim.x >> 5) + wave;
  if (tile >= ntiles) return;  // wave-uniform: EXEC stays all-ones for WMMA

  const int rowBase = tile * 16;
  const int half = lane >> 4;   // 0: lanes 0-15, 1: lanes 16-31
  const int m    = lane & 15;   // A-row owned by this lane
  const int n    = lane & 15;   // B/C column owned by this lane

  // clamp row (duplicate last row in partial tiles; store is guarded below)
  int myRow = rowBase + m;
  myRow = myRow < (nrows - 1) ? myRow : (nrows - 1);
  const float* rowPtr = X + (size_t)myRow * HID;

  // every lane loads from a VALID weight pointer; zero columns >= 2 by mask
  const float* wbase = (n == 0) ? w0 : w1;
  const float  wmask = (n < 2) ? 1.0f : 0.0f;

  floatx8 c = {};  // f32 accumulator, 16x16

  #pragma unroll
  for (int kc = 0; kc < 4; ++kc) {
    const int kbase = kc * 32;

    // A 16x32 bf16 layout (wave32): lanes 0-15 -> K kbase+[0..7] then +[16..23]
    //                               lanes 16-31 -> K kbase+[8..15] then +[24..31]
    const int ka0 = kbase + half * 8;        // 32B-aligned offset
    const int ka1 = kbase + 16 + half * 8;
    floatx8 fa0 = *(const floatx8*)(rowPtr + ka0);
    floatx8 fa1 = *(const floatx8*)(rowPtr + ka1);
    bf16x8 a0 = __builtin_convertvector(fa0, bf16x8);
    bf16x8 a1 = __builtin_convertvector(fa1, bf16x8);
    bf16x16 a = __builtin_shufflevector(a0, a1, 0, 1, 2, 3, 4, 5, 6, 7,
                                        8, 9, 10, 11, 12, 13, 14, 15);

    // B 32x16 bf16 layout: lanes 0-15 hold K kbase+[0..7] then +[16..23] of column n,
    //                      lanes 16-31 hold K kbase+[8..15] then +[24..31] of column n.
    const int kb = kbase + half * 8;         // 32B-aligned offset
    floatx8 fb0 = *(const floatx8*)(wbase + kb);
    floatx8 fb1 = *(const floatx8*)(wbase + kb + 16);
    fb0 *= wmask;
    fb1 *= wmask;
    bf16x8 b0 = __builtin_convertvector(fb0, bf16x8);
    bf16x8 b1 = __builtin_convertvector(fb1, bf16x8);
    bf16x16 b = __builtin_shufflevector(b0, b1, 0, 1, 2, 3, 4, 5, 6, 7,
                                        8, 9, 10, 11, 12, 13, 14, 15);

    c = __builtin_amdgcn_wmma_f32_16x16x32_bf16(
            /*neg_a=*/false, a, /*neg_b=*/false, b,
            /*c_mod=*/(short)0, c, /*reuse_a=*/false, /*reuse_b=*/false);
  }

  // C layout: lanes 0-15 VGPR r -> M=r; lanes 16-31 VGPR r -> M=8+r; N = lane&15
  if (n == 0) {
    #pragma unroll
    for (int r = 0; r < 8; ++r) {
      int row = rowBase + half * 8 + r;
      if (row < nrows) s0[row] = c[r];
    }
  } else if (n == 1 && s1 != (float*)0) {
    #pragma unroll
    for (int r = 0; r < 8; ++r) {
      int row = rowBase + half * 8 + r;
      if (row < nrows) s1[row] = c[r];
    }
  }
}

// ---------------------------------------------------------------------------
// Phase 2: per-edge logits -> segment max (encoded u32 atomicMax).
// ---------------------------------------------------------------------------
__global__ void edge_max_kernel(const int* __restrict__ head,
                                const int* __restrict__ tail,
                                const int* __restrict__ rel,
                                const float* __restrict__ s_h,
                                const float* __restrict__ s_t,
                                const float* __restrict__ s_r,
                                unsigned* __restrict__ m_h,
                                unsigned* __restrict__ m_t, int E) {
  int e = blockIdx.x * blockDim.x + threadIdx.x;
  if (e >= E) return;
  int h = head[e], t = tail[e], r = rel[e];
  float sr = s_r[r];
  atomicMax(&m_h[h], enc_f(lrelu(s_h[h] + sr)));
  atomicMax(&m_t[t], enc_f(lrelu(s_t[t] + sr)));
}

// ---------------------------------------------------------------------------
// Phase 3: per-edge exp(logit - max) -> segment sums.
// ---------------------------------------------------------------------------
__global__ void edge_sum_kernel(const int* __restrict__ head,
                                const int* __restrict__ tail,
                                const int* __restrict__ rel,
                                const float* __restrict__ s_h,
                                const float* __restrict__ s_t,
                                const float* __restrict__ s_r,
                                const unsigned* __restrict__ m_h,
                                const unsigned* __restrict__ m_t,
                                float* __restrict__ sum_h,
                                float* __restrict__ sum_t, int E) {
  int e = blockIdx.x * blockDim.x + threadIdx.x;
  if (e >= E) return;
  int h = head[e], t = tail[e], r = rel[e];
  float sr = s_r[r];
  fatomicAdd(&sum_h[h], __expf(lrelu(s_h[h] + sr) - dec_f(m_h[h])));
  fatomicAdd(&sum_t[t], __expf(lrelu(s_t[t] + sr) - dec_f(m_t[t])));
}

// ---------------------------------------------------------------------------
// Phase 4: one wave per edge; scatter alpha * x_r[rel] into out[head | tail].
// out[:, 0:128] <- head aggregation, out[:, 128:256] <- tail aggregation.
// ---------------------------------------------------------------------------
__global__ void edge_scatter_kernel(const int* __restrict__ head,
                                    const int* __restrict__ tail,
                                    const int* __restrict__ rel,
                                    const float* __restrict__ s_h,
                                    const float* __restrict__ s_t,
                                    const float* __restrict__ s_r,
                                    const unsigned* __restrict__ m_h,
                                    const unsigned* __restrict__ m_t,
                                    const float* __restrict__ sum_h,
                                    const float* __restrict__ sum_t,
                                    const float* __restrict__ x_r,
                                    float* __restrict__ out, int E) {
  int gwave = (int)((blockIdx.x * blockDim.x + threadIdx.x) >> 5);
  int lane  = threadIdx.x & 31;
  if (gwave >= E) return;
  int e = gwave;
  int h = head[e], t = tail[e], r = rel[e];
  float sr = s_r[r];
  // same-address scalar loads across the wave -> broadcast from cache
  float ch = __expf(lrelu(s_h[h] + sr) - dec_f(m_h[h])) / (sum_h[h] + 1e-16f);
  float ct = __expf(lrelu(s_t[t] + sr) - dec_f(m_t[t])) / (sum_t[t] + 1e-16f);

  const float* xr = x_r + (size_t)r * HID;           // L2-resident (512 KB total)
  float* oh = out + (size_t)h * (2 * HID);
  float* ot = out + (size_t)t * (2 * HID) + HID;

  #pragma unroll
  for (int jj = 0; jj < HID / 32; ++jj) {
    int j = lane + jj * 32;
    float v = xr[j];
    fatomicAdd(oh + j, ch * v);
    fatomicAdd(ot + j, ct * v);
  }
}

// ---------------------------------------------------------------------------
extern "C" void kernel_launch(void* const* d_in, const int* in_sizes, int n_in,
                              void* d_out, int out_size, void* d_ws, size_t ws_size,
                              hipStream_t stream) {
  const float* x_e        = (const float*)d_in[0];
  const float* x_r        = (const float*)d_in[1];
  const int*   edge_index = (const int*)d_in[2];
  const int*   rel        = (const int*)d_in[3];
  const float* w_h        = (const float*)d_in[4];
  const float* w_t        = (const float*)d_in[5];
  const float* w_r        = (const float*)d_in[6];

  const int N_E = in_sizes[0] / HID;
  const int N_R = in_sizes[1] / HID;
  const int E   = in_sizes[3];

  const int* head = edge_index;
  const int* tail = edge_index + E;
  float* out = (float*)d_out;

  // Workspace layout (floats/u32, 4B each):
  //   s_h[N_E] s_t[N_E] s_r[N_R] | m_h[N_E] m_t[N_E] sum_h[N_E] sum_t[N_E]
  char* ws = (char*)d_ws;
  float*    s_h   = (float*)ws;                 ws += (size_t)N_E * 4;
  float*    s_t   = (float*)ws;                 ws += (size_t)N_E * 4;
  float*    s_r   = (float*)ws;                 ws += (size_t)N_R * 4;
  unsigned* m_h   = (unsigned*)ws;              ws += (size_t)N_E * 4;
  unsigned* m_t   = (unsigned*)ws;              ws += (size_t)N_E * 4;
  float*    sum_h = (float*)ws;                 ws += (size_t)N_E * 4;
  float*    sum_t = (float*)ws;                 ws += (size_t)N_E * 4;

  // zero max-encodings (enc 0 < enc(-inf): acts as -inf), sums, and output
  hipMemsetAsync(m_h, 0, (size_t)N_E * 4 * sizeof(unsigned), stream);
  hipMemsetAsync(out, 0, (size_t)N_E * (2 * HID) * sizeof(float), stream);

  // Phase 1: WMMA score GEMVs (4 waves / 128-thread block)
  {
    int ntiles_e = (N_E + 15) / 16;
    int grid_e   = (ntiles_e + 3) / 4;
    scores_wmma_kernel<<<grid_e, 128, 0, stream>>>(x_e, N_E, w_h, w_t, s_h, s_t, ntiles_e);

    int ntiles_r = (N_R + 15) / 16;
    int grid_r   = (ntiles_r + 3) / 4;
    scores_wmma_kernel<<<grid_r, 128, 0, stream>>>(x_r, N_R, w_r, w_r, s_r, (float*)0, ntiles_r);
  }

  // Phase 2: segment max
  edge_max_kernel<<<(E + 255) / 256, 256, 0, stream>>>(
      head, tail, rel, s_h, s_t, s_r, m_h, m_t, E);

  // Phase 3: segment sum of exp
  edge_sum_kernel<<<(E + 255) / 256, 256, 0, stream>>>(
      head, tail, rel, s_h, s_t, s_r, m_h, m_t, sum_h, sum_t, E);

  // Phase 4: weighted scatter (one wave per edge, 8 edges / 256-thread block)
  edge_scatter_kernel<<<(E + 7) / 8, 256, 0, stream>>>(
      head, tail, rel, s_h, s_t, s_r, m_h, m_t, sum_h, sum_t, x_r, out, E);
}
